// AMK_Block_59450937311471
// MI455X (gfx1250) — matrix operand order repaired and checked by reading, not compiled, not run on hardware
//
#include <hip/hip_runtime.h>
#include <math.h>

// ---------------------------------------------------------------------------
// MI455X (gfx1250) implementation. wave32, WMMA 16x16x32 bf16, f32 accum.
// Reference: LN -> QKV -> relu^2 linear attention -> Wo -> LN -> gated MLP
//            -> depthwise conv -> Wdown, with residuals.
// Weight packing uses the Tensor Data Mover (tensor_load_to_lds + TENSORcnt).
// ---------------------------------------------------------------------------

typedef __bf16 bf16_t;
typedef bf16_t v16bf __attribute__((ext_vector_type(16)));
typedef bf16_t v8bf  __attribute__((ext_vector_type(8)));
typedef float  v8f   __attribute__((ext_vector_type(8)));
typedef int    v4i   __attribute__((ext_vector_type(4)));
typedef int    v8i   __attribute__((ext_vector_type(8)));

#define B_     2
#define N_     2048
#define D_     1024
#define H_     16
#define DH_    64
#define INNER_ 4096
#define M_     (B_ * N_)   // 4096 total rows

__device__ __forceinline__ v8f zero8() {
    v8f z;
#pragma unroll
    for (int i = 0; i < 8; ++i) z[i] = 0.0f;
    return z;
}

__device__ __forceinline__ v16bf frag_cat(v8bf lo, v8bf hi) {
    return __builtin_shufflevector(lo, hi, 0, 1, 2, 3, 4, 5, 6, 7,
                                           8, 9, 10, 11, 12, 13, 14, 15);
}

__device__ __forceinline__ v8f wmma_bf16(v16bf a, v16bf b, v8f c) {
    return __builtin_amdgcn_wmma_f32_16x16x32_bf16(
        /*neg_a=*/false, a, /*neg_b=*/false, b,
        /*c_mod=*/(short)0, c, /*reuse_a=*/false, /*reuse_b=*/false);
}

// A fragment: 16x32 (MxK) bf16, A row-major with leading dim lda.
// Per-lane: row = mbase + (lane&15); chunks at k = kbase + lhi*8 and +16.
__device__ __forceinline__ v16bf load_a(const bf16_t* A, int lda, int mbase,
                                        int kbase, int ln15, int lhi) {
    const bf16_t* p = A + (size_t)(mbase + ln15) * lda + kbase + lhi * 8;
    v8bf lo = *(const v8bf*)(p);
    v8bf hi = *(const v8bf*)(p + 16);
    return frag_cat(lo, hi);
}

// B fragment: 32x16 (KxN) bf16, from "packed" layout Bp[n * ldb + k]
// (i.e. column-major weights). Per-lane: col = nbase + (lane&15),
// chunks at k = kbase + lhi*8 and +16.
__device__ __forceinline__ v16bf load_b(const bf16_t* Bp, int ldb, int nbase,
                                        int kbase, int ln15, int lhi) {
    const bf16_t* p = Bp + (size_t)(nbase + ln15) * ldb + kbase + lhi * 8;
    v8bf lo = *(const v8bf*)(p);
    v8bf hi = *(const v8bf*)(p + 16);
    return frag_cat(lo, hi);
}

// ---------------------------------------------------------------------------
// Weight pack: f32 W[K][Nc] row-major  ->  bf16 Wt[Nc][K] (column-major).
// The 32x32 f32 tile is staged to LDS by ONE Tensor-Data-Mover descriptor
// (tensor_load_to_lds), with TDM padding producing a 33-float LDS pitch to
// kill bank conflicts on the transposed read.  Completion via TENSORcnt.
// ---------------------------------------------------------------------------
__global__ void __launch_bounds__(256) wpack_kernel(const float* __restrict__ W,
                                                    bf16_t* __restrict__ Wt,
                                                    int K, int Nc) {
    __shared__ __align__(16) float tile[32][33];  // 33-f32 pitch from TDM pad
    int kb = blockIdx.y * 32, nb = blockIdx.x * 32;
    int tx = threadIdx.x, ty = threadIdx.y;  // 32 x 8

    if (ty == 0) {  // wave 0 issues the TDM transfer (TDM ignores EXEC)
        unsigned lds = (unsigned)(size_t)(&tile[0][0]);  // flat LDS: addr[31:0]
        unsigned long long ga =
            (unsigned long long)(size_t)(W + (size_t)kb * Nc + nb);

        // D# group 0: count=1 | lds_addr | global_addr[56:0] | type=2
        v4i g0;
        g0[0] = 1;
        g0[1] = (int)lds;
        g0[2] = (int)(ga & 0xffffffffu);
        g0[3] = (int)((ga >> 32) & 0x01ffffffu) | (2 << 30);

        // D# group 1: data_size=4B (2), pad_enable, pad_interval=4 (32 DW),
        // pad_amount=0 (1 DW); tensor_dim0=Nc, tensor_dim1=K,
        // tile_dim0=tile_dim1=32, tensor_dim0_stride=Nc.
        v8i g1;
        g1[0] = (2 << 16) | (1 << 20) | (4 << 22);
        g1[1] = (Nc & 0xffff) << 16;                       // tensor_dim0[15:0]
        g1[2] = ((Nc >> 16) & 0xffff) | ((K & 0xffff) << 16);
        g1[3] = ((K >> 16) & 0xffff) | (32 << 16);         // tile_dim0 = 32
        g1[4] = 32;                                        // tile_dim1 = 32
        g1[5] = Nc;                                        // dim0_stride[31:0]
        g1[6] = 0;
        g1[7] = 0;

        asm volatile("tensor_load_to_lds %0, %1"
                     :
                     : "s"(g0), "s"(g1)
                     : "memory");
        __builtin_amdgcn_s_wait_tensorcnt(0);
    }
    __syncthreads();

#pragma unroll
    for (int i = ty; i < 32; i += 8)
        Wt[(size_t)(nb + i) * K + kb + tx] = (bf16_t)tile[tx][i];
}

// ---------------------------------------------------------------------------
// LayerNorm (+ optional residual add), bf16 output.  One block per row.
// ---------------------------------------------------------------------------
__global__ void __launch_bounds__(256) ln_kernel(const float* __restrict__ x,
                                                 const float* __restrict__ addx,
                                                 const float* __restrict__ g,
                                                 const float* __restrict__ bb,
                                                 bf16_t* __restrict__ out) {
    __shared__ float s1[256];
    __shared__ float s2[256];
    int row = blockIdx.x;
    int tid = threadIdx.x;
    const float* xr = x + (size_t)row * D_;
    float lsum = 0.0f, lsq = 0.0f;
    for (int i = tid; i < D_; i += 256) {
        float v = xr[i];
        lsum += v;
        lsq += v * v;
    }
    s1[tid] = lsum;
    s2[tid] = lsq;
    __syncthreads();
    for (int s = 128; s > 0; s >>= 1) {
        if (tid < s) { s1[tid] += s1[tid + s]; s2[tid] += s2[tid + s]; }
        __syncthreads();
    }
    float mu = s1[0] * (1.0f / D_);
    float var = s2[0] * (1.0f / D_) - mu * mu;
    float rs = rsqrtf(var + 1e-5f);
    for (int i = tid; i < D_; i += 256) {
        float v = (xr[i] - mu) * rs * g[i] + bb[i];
        if (addx) v += addx[(size_t)row * D_ + i];
        out[(size_t)row * D_ + i] = (bf16_t)v;
    }
}

// ---------------------------------------------------------------------------
// Generic WMMA GEMM: C(MxN) = A(MxK, bf16 rowmajor) * Bp(NxK, bf16 packed).
// Block tile 128x64, 8 waves (4x2), wave tile 32x32 (2x2 WMMA tiles).
// Epilogue variants fuse activation / residual / scatter.
// ---------------------------------------------------------------------------
#define EPI_QK   0   // phi = elu(v)+1 -> bf16 (B,H,N,DH)          [ep_bf]
#define EPI_V    1   // v -> f32 (B,H,N,DH) [ep_out] and bf16 (B,H,DH,N) [ep_bf]
#define EPI_O    2   // out = ep_in + softplus(*dt) * v -> f32      [ep_out]
#define EPI_DOWN 3   // out = ep_in + v -> f32                      [ep_out]

template <int EPI>
__global__ void __launch_bounds__(256) gemm_bf16_kernel(
    const bf16_t* __restrict__ A, const bf16_t* __restrict__ Bp,
    int M, int Ncols, int K,
    const float* __restrict__ ep_in, float* __restrict__ ep_out,
    bf16_t* __restrict__ ep_bf, const float* __restrict__ dt_ptr) {
    int lane = threadIdx.x & 31, wave = threadIdx.x >> 5;
    int ln15 = lane & 15, lhi = lane >> 4;
    int wm = wave & 3, wn = wave >> 2;
    int Mb = blockIdx.x * 128 + wm * 32;
    int Nb = blockIdx.y * 64 + wn * 32;

    v8f acc[2][2];
#pragma unroll
    for (int i = 0; i < 2; ++i)
#pragma unroll
        for (int j = 0; j < 2; ++j) acc[i][j] = zero8();

    for (int kb = 0; kb < K; kb += 32) {
        if (kb + 32 < K) {
            __builtin_prefetch(A + (size_t)(Mb + ln15) * K + kb + 32, 0, 1);
            __builtin_prefetch(Bp + (size_t)(Nb + ln15) * K + kb + 32, 0, 1);
        }
        v16bf a0 = load_a(A, K, Mb, kb, ln15, lhi);
        v16bf a1 = load_a(A, K, Mb + 16, kb, ln15, lhi);
        v16bf b0 = load_b(Bp, K, Nb, kb, ln15, lhi);
        v16bf b1 = load_b(Bp, K, Nb + 16, kb, ln15, lhi);
        acc[0][0] = wmma_bf16(a0, b0, acc[0][0]);
        acc[0][1] = wmma_bf16(a0, b1, acc[0][1]);
        acc[1][0] = wmma_bf16(a1, b0, acc[1][0]);
        acc[1][1] = wmma_bf16(a1, b1, acc[1][1]);
    }

    float sp = 0.0f;
    if (EPI == EPI_O) {
        float dtv = *dt_ptr;
        sp = logf(1.0f + expf(dtv));  // softplus, uniform across threads
    }

#pragma unroll
    for (int ti = 0; ti < 2; ++ti)
#pragma unroll
        for (int tj = 0; tj < 2; ++tj)
#pragma unroll
            for (int r = 0; r < 8; ++r) {
                int row = Mb + ti * 16 + lhi * 8 + r;
                int col = Nb + tj * 16 + ln15;
                float v = acc[ti][tj][r];
                size_t oidx = (size_t)row * Ncols + col;
                if (EPI == EPI_QK) {
                    float phi = v > 0.0f ? v + 1.0f : __expf(v);
                    int b = row >> 11, n = row & (N_ - 1);
                    int h = col >> 6, dh = col & 63;
                    ep_bf[((size_t)(b * H_ + h) * N_ + n) * DH_ + dh] = (bf16_t)phi;
                } else if (EPI == EPI_V) {
                    int b = row >> 11, n = row & (N_ - 1);
                    int h = col >> 6, dh = col & 63;
                    size_t bh = (size_t)(b * H_ + h);
                    ep_out[(bh * N_ + n) * DH_ + dh] = v;
                    ep_bf[(bh * DH_ + dh) * N_ + n] = (bf16_t)v;
                } else if (EPI == EPI_O) {
                    ep_out[oidx] = ep_in[oidx] + sp * v;
                } else {  // EPI_DOWN
                    ep_out[oidx] = ep_in[oidx] + v;
                }
            }
}

// ---------------------------------------------------------------------------
// Linear attention with relu^2 kernel.  Per wave: 16 query rows, full DH=64.
// S = PhiQ*PhiK^T tile -> relu^2 -> (LDS relayout) -> Attr += S*V,
// Norm += S * ones (free 5th output tile = rowsum).
// Epilogue: m = Attr/(Norm+1) - V written to (B,N,D) bf16.
// ---------------------------------------------------------------------------
__global__ void __launch_bounds__(256) attn_kernel(
    const bf16_t* __restrict__ PhiQ,  // (B,H,N,DH) bf16
    const bf16_t* __restrict__ PhiK,  // (B,H,N,DH) bf16
    const bf16_t* __restrict__ Vt,    // (B,H,DH,N) bf16
    const float* __restrict__ Vf,     // (B,H,N,DH) f32
    bf16_t* __restrict__ Mout)        // (B,N,D) bf16
{
    __shared__ __align__(16) bf16_t stag[8][16][32];
    int lane = threadIdx.x & 31, wave = threadIdx.x >> 5;
    int ln15 = lane & 15, lhi = lane >> 4;
    int bh = blockIdx.y;  // b*H + h
    int qbase = blockIdx.x * 128 + wave * 16;

    const bf16_t* Qp  = PhiQ + (size_t)bh * N_ * DH_;
    const bf16_t* Kp  = PhiK + (size_t)bh * N_ * DH_;
    const bf16_t* Vtp = Vt + (size_t)bh * DH_ * N_;

    // Query A-fragments are loop invariant (K = DH = 64 -> two frags).
    v16bf aq0 = load_a(Qp, DH_, qbase, 0, ln15, lhi);
    v16bf aq1 = load_a(Qp, DH_, qbase, 32, ln15, lhi);

    v16bf ones;
#pragma unroll
    for (int i = 0; i < 16; ++i) ones[i] = (bf16_t)1.0f;

    v8f acc[5];
#pragma unroll
    for (int t = 0; t < 5; ++t) acc[t] = zero8();

    for (int kb = 0; kb < N_; kb += 32) {
        // --- S tiles: 16 queries x 32 keys, contraction over DH=64 ---
        v8f s0 = zero8(), s1 = zero8();
        v16bf bk00 = load_b(Kp, DH_, kb, 0, ln15, lhi);
        v16bf bk01 = load_b(Kp, DH_, kb, 32, ln15, lhi);
        v16bf bk10 = load_b(Kp, DH_, kb + 16, 0, ln15, lhi);
        v16bf bk11 = load_b(Kp, DH_, kb + 16, 32, ln15, lhi);
        s0 = wmma_bf16(aq0, bk00, s0);
        s0 = wmma_bf16(aq1, bk01, s0);
        s1 = wmma_bf16(aq0, bk10, s1);
        s1 = wmma_bf16(aq1, bk11, s1);

        // relu^2 and stage through LDS to re-layout C -> A fragment.
#pragma unroll
        for (int r = 0; r < 8; ++r) {
            float v0 = s0[r]; v0 = v0 > 0.0f ? v0 * v0 : 0.0f;
            float v1 = s1[r]; v1 = v1 > 0.0f ? v1 * v1 : 0.0f;
            stag[wave][lhi * 8 + r][ln15]      = (bf16_t)v0;
            stag[wave][lhi * 8 + r][16 + ln15] = (bf16_t)v1;
        }
        v16bf sa = load_a(&stag[wave][0][0], 32, 0, 0, ln15, lhi);

        // --- Attr += S * V   (4 dh tiles) ; Norm += S * 1 ---
#pragma unroll
        for (int t = 0; t < 4; ++t) {
            v16bf bv = load_b(Vtp, N_, t * 16, kb, ln15, lhi);
            acc[t] = wmma_bf16(sa, bv, acc[t]);
        }
        acc[4] = wmma_bf16(sa, ones, acc[4]);
    }

    int b = bh / H_, h = bh % H_;
#pragma unroll
    for (int r = 0; r < 8; ++r) {
        int q = qbase + lhi * 8 + r;
        float norm = acc[4][r] + 1.0f;
#pragma unroll
        for (int t = 0; t < 4; ++t) {
            int dh = t * 16 + ln15;
            float attr = acc[t][r];
            float vv = Vf[((size_t)bh * N_ + q) * DH_ + dh];
            float m = attr / norm - vv;
            Mout[(size_t)(b * N_ + q) * D_ + h * DH_ + dh] = (bf16_t)m;
        }
    }
}

// ---------------------------------------------------------------------------
// Gated-up GEMM: computes G and U halves of Qn2 @ Wup simultaneously and
// writes Hf = silu(G) * U as bf16 (B,N,INNER).  Avoids a 64MB GU tensor.
// ---------------------------------------------------------------------------
__global__ void __launch_bounds__(256) gemm_gate_kernel(
    const bf16_t* __restrict__ A,    // (M, D) bf16
    const bf16_t* __restrict__ Bp,   // (2*INNER, D) packed bf16
    bf16_t* __restrict__ Hf)         // (M, INNER) bf16
{
    int lane = threadIdx.x & 31, wave = threadIdx.x >> 5;
    int ln15 = lane & 15, lhi = lane >> 4;
    int wm = wave & 3, wn = wave >> 2;
    int Mb = blockIdx.x * 128 + wm * 32;
    int Nb = blockIdx.y * 64 + wn * 32;

    v8f g[2][2], u[2][2];
#pragma unroll
    for (int i = 0; i < 2; ++i)
#pragma unroll
        for (int j = 0; j < 2; ++j) { g[i][j] = zero8(); u[i][j] = zero8(); }

    for (int kb = 0; kb < D_; kb += 32) {
        v16bf a0 = load_a(A, D_, Mb, kb, ln15, lhi);
        v16bf a1 = load_a(A, D_, Mb + 16, kb, ln15, lhi);
        v16bf bg0 = load_b(Bp, D_, Nb, kb, ln15, lhi);
        v16bf bg1 = load_b(Bp, D_, Nb + 16, kb, ln15, lhi);
        v16bf bu0 = load_b(Bp, D_, Nb + INNER_, kb, ln15, lhi);
        v16bf bu1 = load_b(Bp, D_, Nb + 16 + INNER_, kb, ln15, lhi);
        g[0][0] = wmma_bf16(a0, bg0, g[0][0]);
        g[0][1] = wmma_bf16(a0, bg1, g[0][1]);
        g[1][0] = wmma_bf16(a1, bg0, g[1][0]);
        g[1][1] = wmma_bf16(a1, bg1, g[1][1]);
        u[0][0] = wmma_bf16(a0, bu0, u[0][0]);
        u[0][1] = wmma_bf16(a0, bu1, u[0][1]);
        u[1][0] = wmma_bf16(a1, bu0, u[1][0]);
        u[1][1] = wmma_bf16(a1, bu1, u[1][1]);
    }

#pragma unroll
    for (int ti = 0; ti < 2; ++ti)
#pragma unroll
        for (int tj = 0; tj < 2; ++tj)
#pragma unroll
            for (int r = 0; r < 8; ++r) {
                int row = Mb + ti * 16 + lhi * 8 + r;
                int col = Nb + tj * 16 + ln15;
                float gv = g[ti][tj][r];
                float uv = u[ti][tj][r];
                float hf = gv / (1.0f + __expf(-gv)) * uv;  // silu(g)*u
                Hf[(size_t)row * INNER_ + col] = (bf16_t)hf;
            }
}

// ---------------------------------------------------------------------------
// Depthwise conv over sequence dim, KS=3, pad=1, per-channel weights.
// ---------------------------------------------------------------------------
__global__ void __launch_bounds__(256) conv_kernel(const bf16_t* __restrict__ Hf,
                                                   const float* __restrict__ cw,
                                                   bf16_t* __restrict__ Hc) {
    size_t idx = (size_t)blockIdx.x * 256 + threadIdx.x;
    int c = (int)(idx & (INNER_ - 1));
    int row = (int)(idx >> 12);  // INNER_ == 4096 == 2^12
    int b = row >> 11, n = row & (N_ - 1);
    float accv = 0.0f;
#pragma unroll
    for (int t = 0; t < 3; ++t) {
        int nn = n - 1 + t;
        if (nn >= 0 && nn < N_)
            accv += (float)Hf[(size_t)(b * N_ + nn) * INNER_ + c] * cw[c * 3 + t];
    }
    Hc[idx] = (bf16_t)accv;
}

// ---------------------------------------------------------------------------
// Host-side orchestration
// ---------------------------------------------------------------------------
extern "C" void kernel_launch(void* const* d_in, const int* in_sizes, int n_in,
                              void* d_out, int out_size, void* d_ws, size_t ws_size,
                              hipStream_t stream) {
    const float* Q_in   = (const float*)d_in[0];
    const float* X      = (const float*)d_in[1];
    const float* Wq     = (const float*)d_in[2];
    const float* Wk     = (const float*)d_in[3];
    const float* Wv     = (const float*)d_in[4];
    const float* Wo     = (const float*)d_in[5];
    const float* Wup    = (const float*)d_in[6];
    const float* conv_w = (const float*)d_in[7];
    const float* Wdown  = (const float*)d_in[8];
    const float* g1     = (const float*)d_in[9];
    const float* b1     = (const float*)d_in[10];
    const float* g2     = (const float*)d_in[11];
    const float* b2     = (const float*)d_in[12];
    const float* dt     = (const float*)d_in[13];

    char* ws = (char*)d_ws;
    size_t off = 0;
    auto wsalloc = [&](size_t bytes) -> void* {
        void* p = ws + off;
        off = (off + bytes + 255) & ~(size_t)255;
        return p;
    };

    const size_t MD = (size_t)M_ * D_;
    bf16_t* Hc_bf = (bf16_t*)wsalloc(MD * 2);
    bf16_t* Wq_t  = (bf16_t*)wsalloc((size_t)D_ * D_ * 2);
    bf16_t* Wk_t  = (bf16_t*)wsalloc((size_t)D_ * D_ * 2);
    bf16_t* Wv_t  = (bf16_t*)wsalloc((size_t)D_ * D_ * 2);
    bf16_t* Wo_t  = (bf16_t*)wsalloc((size_t)D_ * D_ * 2);
    bf16_t* Wup_t = (bf16_t*)wsalloc((size_t)D_ * 2 * INNER_ * 2);
    bf16_t* Wdn_t = (bf16_t*)wsalloc((size_t)INNER_ * D_ * 2);
    bf16_t* PhiQ  = (bf16_t*)wsalloc(MD * 2);
    bf16_t* PhiK  = (bf16_t*)wsalloc(MD * 2);
    float*  Vf    = (float*)wsalloc(MD * 4);
    bf16_t* Vt    = (bf16_t*)wsalloc(MD * 2);
    bf16_t* Mb    = (bf16_t*)wsalloc(MD * 2);
    float*  Qi    = (float*)wsalloc(MD * 4);
    bf16_t* Qn2   = (bf16_t*)wsalloc(MD * 2);
    bf16_t* Hfb   = (bf16_t*)wsalloc((size_t)M_ * INNER_ * 2);
    bf16_t* Hcv   = (bf16_t*)wsalloc((size_t)M_ * INNER_ * 2);

    dim3 tp(32, 8);
    // Pack weights to bf16 column-major (TDM-staged tiles)
    wpack_kernel<<<dim3(D_ / 32, D_ / 32), tp, 0, stream>>>(Wq, Wq_t, D_, D_);
    wpack_kernel<<<dim3(D_ / 32, D_ / 32), tp, 0, stream>>>(Wk, Wk_t, D_, D_);
    wpack_kernel<<<dim3(D_ / 32, D_ / 32), tp, 0, stream>>>(Wv, Wv_t, D_, D_);
    wpack_kernel<<<dim3(D_ / 32, D_ / 32), tp, 0, stream>>>(Wo, Wo_t, D_, D_);
    wpack_kernel<<<dim3((2 * INNER_) / 32, D_ / 32), tp, 0, stream>>>(Wup, Wup_t, D_, 2 * INNER_);
    wpack_kernel<<<dim3(D_ / 32, INNER_ / 32), tp, 0, stream>>>(Wdown, Wdn_t, INNER_, D_);

    // Hc = LN(Q_in) + X
    ln_kernel<<<M_, 256, 0, stream>>>(Q_in, X, g1, b1, Hc_bf);

    dim3 gD(M_ / 128, D_ / 64);
    // PhiQ/PhiK = elu(Hc@W)+1 ; V (f32 + transposed bf16)
    gemm_bf16_kernel<EPI_QK><<<gD, 256, 0, stream>>>(Hc_bf, Wq_t, M_, D_, D_,
                                                     nullptr, nullptr, PhiQ, nullptr);
    gemm_bf16_kernel<EPI_QK><<<gD, 256, 0, stream>>>(Hc_bf, Wk_t, M_, D_, D_,
                                                     nullptr, nullptr, PhiK, nullptr);
    gemm_bf16_kernel<EPI_V><<<gD, 256, 0, stream>>>(Hc_bf, Wv_t, M_, D_, D_,
                                                    nullptr, Vf, Vt, nullptr);

    // m = Attr/Norm - V, stored (B,N,D)
    attn_kernel<<<dim3(N_ / 128, B_ * H_), 256, 0, stream>>>(PhiQ, PhiK, Vt, Vf, Mb);

    // Q_interact = Q_in + softplus(dt) * (m @ Wo)
    gemm_bf16_kernel<EPI_O><<<gD, 256, 0, stream>>>(Mb, Wo_t, M_, D_, D_,
                                                    Q_in, Qi, nullptr, dt);

    // Qn2 = LN(Q_interact)
    ln_kernel<<<M_, 256, 0, stream>>>(Qi, nullptr, g2, b2, Qn2);

    // Hf = silu(G) * U
    gemm_gate_kernel<<<dim3(M_ / 128, INNER_ / 64), 256, 0, stream>>>(Qn2, Wup_t, Hfb);

    // depthwise conv along sequence
    conv_kernel<<<(unsigned)((size_t)M_ * INNER_ / 256), 256, 0, stream>>>(Hfb, conv_w, Hcv);

    // out = Q_interact + Hconv @ Wdown
    gemm_bf16_kernel<EPI_DOWN><<<gD, 256, 0, stream>>>(Hcv, Wdn_t, M_, D_, INNER_,
                                                       Qi, (float*)d_out, nullptr, nullptr);
}
